// _NonLocalBlockND_21560735826262
// MI455X (gfx1250) — compile-verified, matrix operand order checked by reading, not run on hardware
//
#include <hip/hip_runtime.h>
#include <cstddef>
#include <cstdint>

typedef __attribute__((ext_vector_type(16))) __bf16 v16bf;
typedef __attribute__((ext_vector_type(8)))  __bf16 v8bf;
typedef __attribute__((ext_vector_type(8)))  float  v8f;

constexpr int NB  = 4;
constexpr int NC  = 1024;
constexpr int NCI = 512;
constexpr int NT  = 16, NH = 28, NW = 28;
constexpr int NN  = NT * NH * NW;                 // 12544
constexpr int NM  = (NT/2) * (NH/2) * (NW/2);     // 1568
constexpr float BN_EPS = 1e-5f;

constexpr int BM = 128, BNT = 128, BK = 32, LS = 40;  // LDS row stride (elems), 80B = 5*16B

enum : int { EPI_BF16 = 0, EPI_BIAS_TRANS = 1, EPI_FINAL = 2 };

// ---- CDNA5 async global->LDS staging (ASYNCcnt-tracked DMA), with fallback --
#if defined(__gfx1250__) && __has_builtin(__builtin_amdgcn_global_load_async_to_lds_b128)
#define HAS_ASYNC_LDS 1
typedef int v4i_gnu __attribute__((vector_size(16)));
typedef __attribute__((address_space(1))) v4i_gnu ga_v4i;   // global 128-bit chunk
typedef __attribute__((address_space(3))) v4i_gnu la_v4i;   // LDS 128-bit chunk
#else
#define HAS_ASYNC_LDS 0
#endif

__device__ __forceinline__ void stage16B(const __bf16* gp, __bf16* lp)
{
#if HAS_ASYNC_LDS
    __builtin_amdgcn_global_load_async_to_lds_b128((ga_v4i*)gp, (la_v4i*)lp, 0, 0);
#else
    *(v8bf*)lp = *(const v8bf*)gp;
#endif
}

__device__ __forceinline__ void wait_async_stages()
{
#if HAS_ASYNC_LDS
#if __has_builtin(__builtin_amdgcn_s_wait_asynccnt)
    __builtin_amdgcn_s_wait_asynccnt(0);
#else
    asm volatile("s_wait_asynccnt 0x0" ::: "memory");
#endif
#endif
}

// ---------------------------------------------------------------------------
// Generic bf16 WMMA GEMM:  Out(Mr x Nc) = A(Mr x K) * B(K x Nc)
//   A   : row-major (Mr x K), lda
//   B   : if B_IS_NT -> given as (Nc x K) row-major ("BT" layout), else (K x Nc)
//   PARTIAL: bounds-guard staging/stores (only needed when Nc % 128 != 0)
//   256 threads = 8 waves; wave grid 2(M) x 4(N); wave tile = 64 x 32 = 4x2 WMMA
// ---------------------------------------------------------------------------
template <bool B_IS_NT, bool PARTIAL, int EPI>
__global__ __launch_bounds__(256)
void gemm_bf16_wmma(const __bf16* __restrict__ A, size_t aStride,
                    const __bf16* __restrict__ Bm, size_t bStride,
                    void* __restrict__ Out, size_t oStride,
                    int Mr, int Nc, int K, int lda, int ldb, int ldo,
                    const float* __restrict__ bias,
                    const float* __restrict__ gamma,
                    const float* __restrict__ beta,
                    const float* __restrict__ xres)
{
    __shared__ __bf16 As[BM * LS];
    __shared__ __bf16 Bs[BNT * LS];

    const int tid  = threadIdx.x;
    const int lane = tid & 31;
    const int wave = tid >> 5;
    const int wm   = (wave >> 2) * 64;   // wave row offset in block tile
    const int wn   = (wave & 3) * 32;    // wave col offset in block tile
    const int lr   = lane & 15;
    const int kh   = (lane >> 4) << 3;   // K half select: 0 or 8

    const int rowBase = blockIdx.y * BM;
    const int colBase = blockIdx.x * BNT;
    const __bf16* Ab = A  + (size_t)blockIdx.z * aStride;
    const __bf16* Bb = Bm + (size_t)blockIdx.z * bStride;

    v8f acc[4][2];
    #pragma unroll
    for (int i = 0; i < 4; ++i)
        #pragma unroll
        for (int j = 0; j < 2; ++j)
            #pragma unroll
            for (int r = 0; r < 8; ++r) acc[i][j][r] = 0.0f;

    for (int kt = 0; kt < K; kt += BK) {
        // ---- stage A tile (128 x 32) into As[row][k], 16B chunks
        #pragma unroll
        for (int i = 0; i < 2; ++i) {
            int c  = tid + i * 256;
            int r  = c >> 2;
            int kc = (c & 3) << 3;
            if (PARTIAL) {
                v8bf v;
                int gr = rowBase + r;
                if (gr < Mr) {
                    v = *(const v8bf*)(Ab + (size_t)gr * lda + kt + kc);
                } else {
                    #pragma unroll
                    for (int j = 0; j < 8; ++j) v[j] = (__bf16)0.0f;
                }
                *(v8bf*)(As + r * LS + kc) = v;
            } else {
                stage16B(Ab + (size_t)(rowBase + r) * lda + kt + kc, As + r * LS + kc);
            }
        }
        // ---- stage B tile into Bs[col][k]
        if constexpr (B_IS_NT) {
            #pragma unroll
            for (int i = 0; i < 2; ++i) {
                int c  = tid + i * 256;
                int r  = c >> 2;
                int kc = (c & 3) << 3;
                if (PARTIAL) {
                    v8bf v;
                    int gc = colBase + r;
                    if (gc < Nc) {
                        v = *(const v8bf*)(Bb + (size_t)gc * ldb + kt + kc);
                    } else {
                        #pragma unroll
                        for (int j = 0; j < 8; ++j) v[j] = (__bf16)0.0f;
                    }
                    *(v8bf*)(Bs + r * LS + kc) = v;
                } else {
                    stage16B(Bb + (size_t)(colBase + r) * ldb + kt + kc, Bs + r * LS + kc);
                }
            }
        } else {
            // B given as (K x Nc): transpose into Bs[col][k] (scalar LDS writes)
            #pragma unroll
            for (int i = 0; i < 2; ++i) {
                int c   = tid + i * 256;
                int kr  = c >> 4;           // 0..31
                int col = (c & 15) << 3;    // 0..120
                v8bf v;
                if (!PARTIAL || ((colBase + col) < Nc && (kt + kr) < K)) {
                    v = *(const v8bf*)(Bb + (size_t)(kt + kr) * ldb + colBase + col);
                } else {
                    #pragma unroll
                    for (int j = 0; j < 8; ++j) v[j] = (__bf16)0.0f;
                }
                #pragma unroll
                for (int j = 0; j < 8; ++j) Bs[(col + j) * LS + kr] = v[j];
            }
        }
        // prefetch next A tile into cache while we compute on this one
        if (kt + BK < K) {
            int pr = rowBase + (tid >> 2);
            if (pr < Mr)
                __builtin_prefetch(Ab + (size_t)pr * lda + kt + BK, 0, 1);
        }
        wait_async_stages();
        __syncthreads();

        // ---- fragments from LDS (A-style striping for both operands)
        v16bf afr[4], bfr[2];
        #pragma unroll
        for (int mt = 0; mt < 4; ++mt) {
            const __bf16* p = As + (wm + mt * 16 + lr) * LS + kh;
            v8bf lo = *(const v8bf*)p;
            v8bf hi = *(const v8bf*)(p + 16);
            #pragma unroll
            for (int j = 0; j < 8; ++j) { afr[mt][j] = lo[j]; afr[mt][8 + j] = hi[j]; }
        }
        #pragma unroll
        for (int nt = 0; nt < 2; ++nt) {
            const __bf16* p = Bs + (wn + nt * 16 + lr) * LS + kh;
            v8bf lo = *(const v8bf*)p;
            v8bf hi = *(const v8bf*)(p + 16);
            #pragma unroll
            for (int j = 0; j < 8; ++j) { bfr[nt][j] = lo[j]; bfr[nt][8 + j] = hi[j]; }
        }
        #pragma unroll
        for (int mt = 0; mt < 4; ++mt)
            #pragma unroll
            for (int nt = 0; nt < 2; ++nt)
                acc[mt][nt] = __builtin_amdgcn_wmma_f32_16x16x32_bf16(
                    false, afr[mt], false, bfr[nt], (short)0, acc[mt][nt], false, false);
        __syncthreads();
    }

    // ---- epilogue. C/D layout: VGPR r -> row r + 8*(lane>>4), col = lane&15
    #pragma unroll
    for (int mt = 0; mt < 4; ++mt) {
        #pragma unroll
        for (int nt = 0; nt < 2; ++nt) {
            #pragma unroll
            for (int r = 0; r < 8; ++r) {
                int row = rowBase + wm + mt * 16 + r + ((lane >> 4) << 3);
                int col = colBase + wn + nt * 16 + (lane & 15);
                if (!PARTIAL || (row < Mr && col < Nc)) {
                    float v = acc[mt][nt][r];
                    if constexpr (EPI == EPI_BF16) {
                        ((__bf16*)Out)[(size_t)blockIdx.z * oStride + (size_t)row * ldo + col] =
                            (__bf16)v;
                    } else if constexpr (EPI == EPI_BIAS_TRANS) {
                        v += bias[row];
                        ((__bf16*)Out)[(size_t)blockIdx.z * oStride + (size_t)col * ldo + row] =
                            (__bf16)v;
                    } else {
                        // (v + w_b[c]) * gamma[c]/sqrt(1+eps) + beta[c] + x[b][c][n]
                        float s = gamma[col] * rsqrtf(1.0f + BN_EPS);
                        size_t idx = (size_t)blockIdx.z * oStride + (size_t)col * ldo + row;
                        ((float*)Out)[idx] = (v + bias[col]) * s + beta[col] + xres[idx];
                    }
                }
            }
        }
    }
}

// ---------------------------------------------------------------------------
// fp32 -> bf16 elementwise cast (weights)
// ---------------------------------------------------------------------------
__global__ __launch_bounds__(256)
void cast_f32_bf16(const float* __restrict__ src, __bf16* __restrict__ dst, int n)
{
    for (int i = blockIdx.x * 256 + threadIdx.x; i < n; i += gridDim.x * 256)
        dst[i] = (__bf16)src[i];
}

// ---------------------------------------------------------------------------
// x (B, C, N) fp32 -> xT (B, N, C) bf16, 32x32 LDS tile transpose
// ---------------------------------------------------------------------------
__global__ __launch_bounds__(256)
void transpose_cast_x(const float* __restrict__ x, __bf16* __restrict__ xT)
{
    __shared__ __bf16 t[32][33];
    const int tx = threadIdx.x & 31, ty = threadIdx.x >> 5;
    const int n0 = blockIdx.x * 32, c0 = blockIdx.y * 32, b = blockIdx.z;
    const float* xb = x + (size_t)b * NC * NN;
    #pragma unroll
    for (int j = 0; j < 4; ++j)
        t[ty + j * 8][tx] = (__bf16)xb[(size_t)(c0 + ty + j * 8) * NN + n0 + tx];
    __syncthreads();
    __bf16* ob = xT + (size_t)b * NN * NC;
    #pragma unroll
    for (int j = 0; j < 4; ++j)
        ob[(size_t)(n0 + ty + j * 8) * NC + c0 + tx] = t[tx][ty + j * 8];
}

// ---------------------------------------------------------------------------
// 2x2x2 max pool over the spatial index of a (B, N, CI) bf16 tensor -> (B, M, CI)
// ---------------------------------------------------------------------------
__global__ __launch_bounds__(256)
void pool2_rows(const __bf16* __restrict__ in, __bf16* __restrict__ out)
{
    const int total = NB * NM * NCI;
    for (int idx = blockIdx.x * 256 + threadIdx.x; idx < total; idx += gridDim.x * 256) {
        int b   = idx / (NM * NCI);
        int rem = idx - b * (NM * NCI);
        int m   = rem / NCI;
        int ci  = rem - m * NCI;
        int tt = m / ((NH/2) * (NW/2));
        int r2 = m - tt * ((NH/2) * (NW/2));
        int hh = r2 / (NW/2);
        int ww = r2 - hh * (NW/2);
        int n0 = (2 * tt) * (NH * NW) + (2 * hh) * NW + 2 * ww;
        const __bf16* ib = in + (size_t)b * NN * NCI;
        float mx = -3.0e38f;
        #pragma unroll
        for (int dt = 0; dt < 2; ++dt)
            #pragma unroll
            for (int dh = 0; dh < 2; ++dh)
                #pragma unroll
                for (int dw = 0; dw < 2; ++dw) {
                    int n = n0 + dt * (NH * NW) + dh * NW + dw;
                    mx = fmaxf(mx, (float)ib[(size_t)n * NCI + ci]);
                }
        out[(size_t)(b * NM + m) * NCI + ci] = (__bf16)mx;
    }
}

// ---------------------------------------------------------------------------
// in-place row softmax on (B*N, M) bf16 scores, one workgroup per row
// ---------------------------------------------------------------------------
__global__ __launch_bounds__(256)
void softmax_rows(__bf16* __restrict__ f)
{
    __shared__ float red[256];
    __bf16* row = f + (size_t)blockIdx.x * NM;
    const int tid = threadIdx.x;

    float mx = -3.0e38f;
    for (int i = tid; i < NM; i += 256) mx = fmaxf(mx, (float)row[i]);
    red[tid] = mx; __syncthreads();
    for (int s = 128; s > 0; s >>= 1) {
        if (tid < s) red[tid] = fmaxf(red[tid], red[tid + s]);
        __syncthreads();
    }
    mx = red[0]; __syncthreads();

    float sum = 0.0f;
    for (int i = tid; i < NM; i += 256) {
        float e = __expf((float)row[i] - mx);
        sum += e;
        row[i] = (__bf16)e;
    }
    red[tid] = sum; __syncthreads();
    for (int s = 128; s > 0; s >>= 1) {
        if (tid < s) red[tid] += red[tid + s];
        __syncthreads();
    }
    float inv = 1.0f / red[0];
    for (int i = tid; i < NM; i += 256) row[i] = (__bf16)((float)row[i] * inv);
}

// ---------------------------------------------------------------------------
// workspace layout (bytes). f aliases the dead xT/gT/phiT region.
// ---------------------------------------------------------------------------
constexpr size_t SZ_XT   = (size_t)NB * NN * NC  * 2;  // 102,760,448
constexpr size_t SZ_NCI  = (size_t)NB * NN * NCI * 2;  //  51,380,224
constexpr size_t SZ_F    = (size_t)NB * NN * NM  * 2;  // 157,351,936
constexpr size_t SZ_W1   = (size_t)NCI * NC * 2;       //   1,048,576
constexpr size_t SZ_P    = (size_t)NB * NM * NCI * 2;  //   6,422,528

constexpr size_t OFF_XT     = 0;
constexpr size_t OFF_GT     = OFF_XT + SZ_XT;
constexpr size_t OFF_PHIT   = OFF_GT + SZ_NCI;
constexpr size_t OFF_F      = 0;                       // aliases xT+gT+phiT (dead)
constexpr size_t OFF_THETAT = OFF_PHIT + SZ_NCI;
constexpr size_t OFF_WBF    = OFF_THETAT + SZ_NCI;     // 4 weight mats bf16
constexpr size_t OFF_GP     = OFF_WBF + 4 * SZ_W1;
constexpr size_t OFF_PHP    = OFF_GP + SZ_P;
constexpr size_t OFF_Y      = OFF_PHP + SZ_P;
static_assert(SZ_F <= OFF_THETAT, "f alias must fit in dead region");

extern "C" void kernel_launch(void* const* d_in, const int* in_sizes, int n_in,
                              void* d_out, int out_size, void* d_ws, size_t ws_size,
                              hipStream_t stream)
{
    (void)in_sizes; (void)n_in; (void)out_size; (void)ws_size;

    const float* x       = (const float*)d_in[0];
    const float* g_w     = (const float*)d_in[1];
    const float* g_b     = (const float*)d_in[2];
    const float* theta_w = (const float*)d_in[3];
    const float* theta_b = (const float*)d_in[4];
    const float* phi_w   = (const float*)d_in[5];
    const float* phi_b   = (const float*)d_in[6];
    const float* w_w     = (const float*)d_in[7];
    const float* w_b     = (const float*)d_in[8];
    const float* bn_g    = (const float*)d_in[9];
    const float* bn_b    = (const float*)d_in[10];

    char* ws = (char*)d_ws;
    __bf16* xT     = (__bf16*)(ws + OFF_XT);
    __bf16* gT     = (__bf16*)(ws + OFF_GT);
    __bf16* phiT   = (__bf16*)(ws + OFF_PHIT);
    __bf16* thetaT = (__bf16*)(ws + OFF_THETAT);
    __bf16* gwbf   = (__bf16*)(ws + OFF_WBF + 0 * SZ_W1);
    __bf16* thwbf  = (__bf16*)(ws + OFF_WBF + 1 * SZ_W1);
    __bf16* phwbf  = (__bf16*)(ws + OFF_WBF + 2 * SZ_W1);
    __bf16* wwbf   = (__bf16*)(ws + OFF_WBF + 3 * SZ_W1);
    __bf16* gP     = (__bf16*)(ws + OFF_GP);
    __bf16* phP    = (__bf16*)(ws + OFF_PHP);
    __bf16* fbuf   = (__bf16*)(ws + OFF_F);
    __bf16* ybuf   = (__bf16*)(ws + OFF_Y);

    // 1) weights -> bf16
    const int wc = NCI * NC;
    cast_f32_bf16<<<dim3((wc + 255) / 256), 256, 0, stream>>>(g_w,     gwbf,  wc);
    cast_f32_bf16<<<dim3((wc + 255) / 256), 256, 0, stream>>>(theta_w, thwbf, wc);
    cast_f32_bf16<<<dim3((wc + 255) / 256), 256, 0, stream>>>(phi_w,   phwbf, wc);
    cast_f32_bf16<<<dim3((wc + 255) / 256), 256, 0, stream>>>(w_w,     wwbf,  wc);

    // 2) x (B,C,N) -> xT (B,N,C) bf16
    transpose_cast_x<<<dim3(NN / 32, NC / 32, NB), 256, 0, stream>>>(x, xT);

    // 3) conv branches: W(CI x C) * xT'(C x N), stored transposed as (N x CI) bf16
    dim3 gconv(NN / BNT, NCI / BM, NB);
    gemm_bf16_wmma<true, false, EPI_BIAS_TRANS><<<gconv, 256, 0, stream>>>(
        thwbf, 0, xT, (size_t)NN * NC, thetaT, (size_t)NN * NCI,
        NCI, NN, NC, NC, NC, NCI, theta_b, nullptr, nullptr, nullptr);
    gemm_bf16_wmma<true, false, EPI_BIAS_TRANS><<<gconv, 256, 0, stream>>>(
        gwbf, 0, xT, (size_t)NN * NC, gT, (size_t)NN * NCI,
        NCI, NN, NC, NC, NC, NCI, g_b, nullptr, nullptr, nullptr);
    gemm_bf16_wmma<true, false, EPI_BIAS_TRANS><<<gconv, 256, 0, stream>>>(
        phwbf, 0, xT, (size_t)NN * NC, phiT, (size_t)NN * NCI,
        NCI, NN, NC, NC, NC, NCI, phi_b, nullptr, nullptr, nullptr);

    // 4) 2x2x2 max pool: (B,N,CI) -> (B,M,CI)
    const int pc = NB * NM * NCI;
    pool2_rows<<<dim3((pc + 255) / 256), 256, 0, stream>>>(gT, gP);
    pool2_rows<<<dim3((pc + 255) / 256), 256, 0, stream>>>(phiT, phP);

    // 5) f = thetaT (N x CI) * phP' (CI x M), bf16 out; Nc=1568 has a partial tile
    gemm_bf16_wmma<true, true, EPI_BF16><<<dim3((NM + BNT - 1) / BNT, NN / BM, NB), 256, 0, stream>>>(
        thetaT, (size_t)NN * NCI, phP, (size_t)NM * NCI, fbuf, (size_t)NN * NM,
        NN, NM, NCI, NCI, NCI, NM, nullptr, nullptr, nullptr, nullptr);

    // 6) softmax over M, in place
    softmax_rows<<<dim3(NB * NN), 256, 0, stream>>>(fbuf);

    // 7) y = f (N x M) * gP (M x CI), bf16 out (N x CI)
    gemm_bf16_wmma<false, false, EPI_BF16><<<dim3(NCI / BNT, NN / BM, NB), 256, 0, stream>>>(
        fbuf, (size_t)NN * NM, gP, (size_t)NM * NCI, ybuf, (size_t)NN * NCI,
        NN, NCI, NM, NM, NCI, NCI, nullptr, nullptr, nullptr, nullptr);

    // 8) out = y (N x CI) * w_w' (CI x C), fused bias + BN + residual, fp32 (B,C,N)
    gemm_bf16_wmma<true, false, EPI_FINAL><<<dim3(NC / BNT, NN / BM, NB), 256, 0, stream>>>(
        ybuf, (size_t)NN * NCI, wwbf, 0, d_out, (size_t)NC * NN,
        NN, NC, NCI, NCI, NCI, NN, w_b, bn_g, bn_b, x);
}
